// MultiHeadAttention_20083267076124
// MI455X (gfx1250) — matrix operand order repaired and checked by reading, not compiled, not run on hardware
//
#include <hip/hip_runtime.h>

// ---------------------------------------------------------------- types ----
typedef __attribute__((ext_vector_type(16))) __bf16 v16bf;
typedef __attribute__((ext_vector_type(8)))  __bf16 v8bf;
typedef __attribute__((ext_vector_type(4)))  __bf16 v4bf;
typedef __attribute__((ext_vector_type(8)))  float  v8f;

static constexpr int kB    = 2;
static constexpr int kS    = 2048;
static constexpr int kE    = 1024;
static constexpr int kH    = 16;
static constexpr int kDK   = 64;
static constexpr int kM    = kB * kS;   // 4096 rows in all GEMMs

// fp32 -> bf16 round-to-nearest-even
__device__ __forceinline__ __bf16 f2bf(float f) {
  unsigned u = __builtin_bit_cast(unsigned, f);
  u += 0x7FFFu + ((u >> 16) & 1u);
  unsigned short h = (unsigned short)(u >> 16);
  return __builtin_bit_cast(__bf16, h);
}

__device__ __forceinline__ v8f wmma_bf16(v16bf a, v16bf b, v8f c) {
  return __builtin_amdgcn_wmma_f32_16x16x32_bf16(false, a, false, b,
                                                 (short)0, c, false, false);
}

// A-matrix fragment (16x32 bf16): lane holds row r, K = 8*half+{0..7} and 16+8*half+{0..7}
__device__ __forceinline__ v16bf load_a_bf16(const __bf16* rowp, int half) {
  v8bf lo = *(const v8bf*)(rowp + 8 * half);
  v8bf hi = *(const v8bf*)(rowp + 16 + 8 * half);
  v16bf o;
#pragma unroll
  for (int i = 0; i < 8; ++i) { o[i] = lo[i]; o[8 + i] = hi[i]; }
  return o;
}

// B-matrix fragment (32x16 bf16): lane holds column r, K = 8*half+{0..7} and 16+8*half+{0..7}
__device__ __forceinline__ v16bf load_b_bf16(const __bf16* colsrc, int half) {
  v8bf lo = *(const v8bf*)(colsrc + 8 * half);
  v8bf hi = *(const v8bf*)(colsrc + 16 + 8 * half);
  v16bf o;
#pragma unroll
  for (int i = 0; i < 8; ++i) { o[i] = lo[i]; o[8 + i] = hi[i]; }
  return o;
}

// ---------------------------------------------- fp32 -> bf16 pre-conversion
__global__ __launch_bounds__(256)
void mha_cvt_f32_bf16(const float* __restrict__ in, __bf16* __restrict__ out, int n) {
  const int i = (blockIdx.x * blockDim.x + threadIdx.x) * 4;
  if (i >= n) return;
  const float4 f = *(const float4*)(in + i);
  v4bf o;
  o[0] = f2bf(f.x); o[1] = f2bf(f.y); o[2] = f2bf(f.z); o[3] = f2bf(f.w);
  *(v4bf*)(out + i) = o;
}

// ---------------------------------------------------------- projection GEMM
// C[M,N] = A[M,K] @ W[N,K]^T + bias[N]   (A and W both pre-converted bf16)
// Per wave: 32x64 output tile -> 8 WMMAs per 32-wide k-step, pure b128 loads.
// MODE 0: bf16 out, [B,H,S,DK] layout (Q and K projections)
// MODE 1: bf16 out, [B,H,DK,S] layout (V projection, transposed)
// MODE 2: fp32  out, row-major [M,N]  (output projection -> d_out)
template <int MODE>
__global__ __launch_bounds__(128)
void mha_proj_gemm(const __bf16* __restrict__ Ab, const __bf16* __restrict__ Wb,
                   const float* __restrict__ bias, void* __restrict__ Out) {
  constexpr int K = kE, N = kE;
  const int lane = threadIdx.x & 31;
  const int wave = threadIdx.x >> 5;
  const int r    = lane & 15;
  const int half = lane >> 4;
  const int mbase = blockIdx.y * 32;
  const int nbase = blockIdx.x * 256 + wave * 64;

  v8f acc[2][4] = {};

  const __bf16* arow0 = Ab + (size_t)(mbase + r) * K;
  const __bf16* arow1 = Ab + (size_t)(mbase + 16 + r) * K;

  for (int kk = 0; kk < K; kk += 32) {
    // A fragments: rows mbase+r and mbase+16+r
    const v16bf a0 = load_a_bf16(arow0 + kk, half);
    const v16bf a1 = load_a_bf16(arow1 + kk, half);
    // B fragments: 4 n-subtiles of pre-converted weights
    v16bf bfr[4];
#pragma unroll
    for (int t = 0; t < 4; ++t) {
      const __bf16* wrow = Wb + (size_t)(nbase + t * 16 + r) * K + kk;
      bfr[t] = load_b_bf16(wrow, half);
      if (kk + 32 < K) __builtin_prefetch(wrow + 32, 0, 1);   // global_prefetch_b8
    }
#pragma unroll
    for (int t = 0; t < 4; ++t) {
      acc[0][t] = wmma_bf16(a0, bfr[t], acc[0][t]);
      acc[1][t] = wmma_bf16(a1, bfr[t], acc[1][t]);
    }
  }

#pragma unroll
  for (int g = 0; g < 2; ++g) {
#pragma unroll
    for (int t = 0; t < 4; ++t) {
      const int n_col = nbase + t * 16 + r;
      const float bb = bias[n_col];
#pragma unroll
      for (int v = 0; v < 8; ++v) {
        const int row = mbase + g * 16 + v + 8 * half;
        const float val = acc[g][t][v] + bb;
        if constexpr (MODE == 2) {
          ((float*)Out)[(size_t)row * N + n_col] = val;
        } else {
          const int b = row >> 11;        // row / S
          const int s = row & (kS - 1);
          const int h = n_col >> 6;       // n / DK
          const int d = n_col & (kDK - 1);
          size_t idx;
          if constexpr (MODE == 0)
            idx = (((size_t)b * kH + h) * kS + s) * kDK + d;      // [B,H,S,DK]
          else
            idx = (((size_t)b * kH + h) * kDK + d) * kS + s;      // [B,H,DK,S]
          ((__bf16*)Out)[idx] = f2bf(val);
        }
      }
    }
  }
}

// ------------------------------------------------------- flash attention ---
// One wave per (bh, 16-query tile). Causal. Q,K: [B,H,S,DK] bf16, VT: [B,H,DK,S] bf16.
// Output: attn context in [M, E] row-major bf16 (feeds output projection).
__global__ __launch_bounds__(32)
void mha_flash_attn(const __bf16* __restrict__ Q, const __bf16* __restrict__ Kb,
                    const __bf16* __restrict__ VT, __bf16* __restrict__ Obf) {
  __shared__ float lds_s[16 * 32];   // 16 queries x 32 keys score/prob tile

  const int lane = threadIdx.x & 31;
  const int r    = lane & 15;
  const int half = lane >> 4;
  const int bh    = blockIdx.y;                 // b*H + h
  const int qbase = blockIdx.x * 16;
  const int q     = qbase + r;                  // this lane's softmax row

  // Q fragments for d = 0..31 and 32..63
  const __bf16* qrow = Q + ((size_t)bh * kS + qbase + r) * kDK;
  const v16bf aq0 = load_a_bf16(qrow, half);
  const v16bf aq1 = load_a_bf16(qrow + 32, half);

  v8f acc[4] = {};
  float m_r = -__builtin_inff();
  float l_r = 0.0f;

  const int jt_end = (qbase + 15) >> 5;         // last 32-key tile touching causal band
  for (int jt = 0; jt <= jt_end; ++jt) {
    const int j = jt * 32;

    // ---- issue ALL global fragment loads for this tile up front so they overlap
    const __bf16* k0 = Kb + ((size_t)bh * kS + j + r) * kDK;
    const __bf16* k1 = Kb + ((size_t)bh * kS + j + 16 + r) * kDK;
    const v16bf bk00 = load_b_bf16(k0, half);
    const v16bf bk01 = load_b_bf16(k0 + 32, half);
    const v16bf bk10 = load_b_bf16(k1, half);
    const v16bf bk11 = load_b_bf16(k1 + 32, half);
    v16bf bv[4];                                 // V frags: independent of softmax
#pragma unroll
    for (int t = 0; t < 4; ++t)
      bv[t] = load_b_bf16(VT + ((size_t)bh * kDK + t * 16 + r) * kS + j, half);
    if (jt < jt_end) {                           // prefetch next K tile
      __builtin_prefetch(k0 + 32 * kDK, 0, 1);
      __builtin_prefetch(k1 + 32 * kDK, 0, 1);
    }

    // ---- scores: S[16q x 32k] via 4 WMMAs
    v8f s0 = {}, s1 = {};
    s0 = wmma_bf16(aq0, bk00, s0);
    s0 = wmma_bf16(aq1, bk01, s0);
    s1 = wmma_bf16(aq0, bk10, s1);
    s1 = wmma_bf16(aq1, bk11, s1);

    // ---- spill score tiles to LDS (m = v + 8*half, n = r / 16+r)
#pragma unroll
    for (int v = 0; v < 8; ++v) {
      lds_s[(v + 8 * half) * 32 + r]      = s0[v];
      lds_s[(v + 8 * half) * 32 + 16 + r] = s1[v];
    }
    __syncthreads();

    // ---- causal mask (-1e10 BEFORE 1/sqrt(dk), faithful to reference) + online softmax
    // lane (r,half) owns row r, columns half*16 .. half*16+15
    float sc[16];
    float mloc = -__builtin_inff();
#pragma unroll
    for (int c = 0; c < 16; ++c) {
      const int col  = half * 16 + c;
      const int kpos = j + col;
      float sv = lds_s[r * 32 + col];
      sv = (kpos <= q) ? sv : -1.0e10f;
      sv *= 0.125f;                       // 1/sqrt(64)
      sc[c] = sv;
      mloc = fmaxf(mloc, sv);
    }
    mloc = fmaxf(mloc, __shfl_xor(mloc, 16, 32));
    const float mnew  = fmaxf(m_r, mloc);
    const float scale = __expf(m_r - mnew);
    float psum = 0.0f;
#pragma unroll
    for (int c = 0; c < 16; ++c) {
      const float p = __expf(sc[c] - mnew);
      psum += p;
      lds_s[r * 32 + half * 16 + c] = p;  // overwrite scores with probabilities
    }
    psum += __shfl_xor(psum, 16, 32);
    l_r = l_r * scale + psum;
    m_r = mnew;

    // ---- rescale accumulators; row m's factor lives in lane m (both halves agree)
#pragma unroll
    for (int v = 0; v < 8; ++v) {
      const float f = __shfl(scale, v + 8 * half, 32);
      acc[0][v] *= f; acc[1][v] *= f; acc[2][v] *= f; acc[3][v] *= f;
    }
    __syncthreads();

    // ---- P (16q x 32k) A-fragment from LDS, then P @ V with 4 WMMAs (d tiles)
    v16bf pa;
#pragma unroll
    for (int i = 0; i < 8; ++i) {
      pa[i]     = f2bf(lds_s[r * 32 + 8 * half + i]);
      pa[8 + i] = f2bf(lds_s[r * 32 + 16 + 8 * half + i]);
    }
#pragma unroll
    for (int t = 0; t < 4; ++t)
      acc[t] = wmma_bf16(pa, bv[t], acc[t]);
    __syncthreads();
  }

  // ---- normalize by l and store context in [M, E] bf16 (row = b*S + q, col = h*DK + d)
  const int b = bh >> 4;
  const int h = bh & 15;
  const float il = 1.0f / l_r;
#pragma unroll
  for (int t = 0; t < 4; ++t) {
#pragma unroll
    for (int v = 0; v < 8; ++v) {
      const int m = v + 8 * half;
      const float f = __shfl(il, m, 32);
      const size_t row = (size_t)b * kS + qbase + m;
      const int col = h * kDK + t * 16 + r;
      Obf[row * kE + col] = f2bf(acc[t][v] * f);
    }
  }
}

// ------------------------------------------------------------------ host ---
extern "C" void kernel_launch(void* const* d_in, const int* in_sizes, int n_in,
                              void* d_out, int out_size, void* d_ws, size_t ws_size,
                              hipStream_t stream) {
  (void)in_sizes; (void)n_in; (void)out_size; (void)ws_size;

  const float* key   = (const float*)d_in[0];
  const float* value = (const float*)d_in[1];
  const float* query = (const float*)d_in[2];
  // d_in[3] = mask (causal, handled analytically)
  const float* Wk = (const float*)d_in[4];  const float* bk = (const float*)d_in[5];
  const float* Wq = (const float*)d_in[6];  const float* bq = (const float*)d_in[7];
  const float* Wv = (const float*)d_in[8];  const float* bv = (const float*)d_in[9];
  const float* Wo = (const float*)d_in[10]; const float* bo = (const float*)d_in[11];

  const size_t acts = (size_t)kM * kE;      // 4M elements per activation tensor
  const size_t wels = (size_t)kE * kE;      // 1M elements per weight matrix
  __bf16* Qb  = (__bf16*)d_ws;              // Q in [B,H,S,DK]
  __bf16* Kb  = Qb  + acts;                 // K in [B,H,S,DK]
  __bf16* Vt  = Kb  + acts;                 // V in [B,H,DK,S]
  __bf16* Ao  = Vt  + acts;                 // attention context [M, E]
  __bf16* Wqb = Ao  + acts;
  __bf16* Wkb = Wqb + wels;
  __bf16* Wvb = Wkb + wels;
  __bf16* Wob = Wvb + wels;
  __bf16* Xq  = Wob + wels;                 // bf16 copies of activations
  __bf16* Xk  = Xq  + acts;
  __bf16* Xv  = Xk  + acts;                 // 64 MB bf16 workspace total

  // ---- pre-convert weights + activations to bf16 (once; GEMM loops stay pure)
  {
    const int nw = (int)wels;
    dim3 wgrid(nw / (256 * 4));
    mha_cvt_f32_bf16<<<wgrid, 256, 0, stream>>>(Wq, Wqb, nw);
    mha_cvt_f32_bf16<<<wgrid, 256, 0, stream>>>(Wk, Wkb, nw);
    mha_cvt_f32_bf16<<<wgrid, 256, 0, stream>>>(Wv, Wvb, nw);
    mha_cvt_f32_bf16<<<wgrid, 256, 0, stream>>>(Wo, Wob, nw);
    const int na = (int)acts;
    dim3 agrid(na / (256 * 4));
    mha_cvt_f32_bf16<<<agrid, 256, 0, stream>>>(query, Xq, na);
    mha_cvt_f32_bf16<<<agrid, 256, 0, stream>>>(key,   Xk, na);
    mha_cvt_f32_bf16<<<agrid, 256, 0, stream>>>(value, Xv, na);
  }

  dim3 gblk(kE / 256, kM / 32);             // (4, 128)
  dim3 tblk(128);

  mha_proj_gemm<0><<<gblk, tblk, 0, stream>>>(Xq, Wqb, bq, Qb);
  mha_proj_gemm<0><<<gblk, tblk, 0, stream>>>(Xk, Wkb, bk, Kb);
  mha_proj_gemm<1><<<gblk, tblk, 0, stream>>>(Xv, Wvb, bv, Vt);

  dim3 gattn(kS / 16, kB * kH);             // (128, 32)
  mha_flash_attn<<<gattn, dim3(32), 0, stream>>>(Qb, Kb, Vt, Ao);

  mha_proj_gemm<2><<<gblk, tblk, 0, stream>>>(Ao, Wob, bo, (float*)d_out);
}
